// NaiveXNORLinear_89318139887630
// MI455X (gfx1250) — 8-bit matrix operand order repaired and checked by reading, compiled, not run on hardware
//
#include <hip/hip_runtime.h>
#include <hip/hip_bf16.h>

typedef __attribute__((ext_vector_type(16))) int   v16i;
typedef __attribute__((ext_vector_type(8)))  float v8f;
typedef __attribute__((ext_vector_type(4)))  int   i4v;
typedef __attribute__((ext_vector_type(2)))  int   i2v;

__global__ void zero_accum_kernel(float* p) {
    if (threadIdx.x < 2) p[threadIdx.x] = 0.0f;
}

__device__ __forceinline__ unsigned int sign_fp8(float v) {
    return (v > 0.0f) ? 0x38u : ((v < 0.0f) ? 0xB8u : 0x00u);
}

__global__ __launch_bounds__(256) void quantize_abs_kernel(
    const float* __restrict__ in, unsigned char* __restrict__ out8,
    float* __restrict__ accum, long long n4) {
    __shared__ float red[256];
    float s = 0.0f;
    long long idx    = (long long)blockIdx.x * blockDim.x + threadIdx.x;
    long long stride = (long long)gridDim.x * blockDim.x;
    const float4* in4   = (const float4*)in;
    unsigned int* out32 = (unsigned int*)out8;
    for (long long i = idx; i < n4; i += stride) {
        float4 v = in4[i];
        unsigned int b = sign_fp8(v.x)
                       | (sign_fp8(v.y) << 8)
                       | (sign_fp8(v.z) << 16)
                       | (sign_fp8(v.w) << 24);
        out32[i] = b;
        s += fabsf(v.x) + fabsf(v.y) + fabsf(v.z) + fabsf(v.w);
    }
    red[threadIdx.x] = s;
    __syncthreads();
    for (int off = 128; off > 0; off >>= 1) {
        if ((int)threadIdx.x < off) red[threadIdx.x] += red[threadIdx.x + off];
        __syncthreads();
    }
    if (threadIdx.x == 0) atomicAdd(accum, red[0]);
}

__device__ __forceinline__ void async_copy_b128(unsigned lds_off,
                                                const unsigned char* gptr) {
    asm volatile("global_load_async_to_lds_b128 %0, %1, off"
                 :: "v"(lds_off), "v"(gptr) : "memory");
}

__device__ __forceinline__ void wait_async0() {
    asm volatile("s_wait_asynccnt 0x0" ::: "memory");
}

#define LDS_ROW_STRIDE 144
#define TILE_BYTES     (128 * LDS_ROW_STRIDE)
#define BUF_BYTES      (2 * TILE_BYTES)

__device__ __forceinline__ v16i load_a_frag_lds(const unsigned char* row_base, int half) {
    v16i a;
#pragma unroll
    for (int h = 0; h < 2; ++h) {
#pragma unroll
        for (int blk = 0; blk < 4; ++blk) {
            const int koff = h * 64 + blk * 16 + half * 8;
            i2v d = *(const i2v*)(row_base + koff);
            a[h * 8 + blk * 2 + 0] = d.x;
            a[h * 8 + blk * 2 + 1] = d.y;
        }
    }
    return a;
}

__device__ __forceinline__ v16i load_b_frag_lds(const unsigned char* row_base, int half) {
    v16i b;
#pragma unroll
    for (int g = 0; g < 4; ++g) {
        const int koff = g * 32 + half * 16;
        i4v d = *(const i4v*)(row_base + koff);
        b[4 * g + 0] = d.x;
        b[4 * g + 1] = d.y;
        b[4 * g + 2] = d.z;
        b[4 * g + 3] = d.w;
    }
    return b;
}

__global__ __launch_bounds__(256) void xnor_gemm_fp8_wmma_kernel(
    const unsigned char* __restrict__ qx8,
    const unsigned char* __restrict__ qw8,
    const float* __restrict__ bias,
    float* __restrict__ out,
    const float* __restrict__ accum,
    int CIN, int COUT, float inv_xn, float inv_wn) {
    __shared__ __align__(16) unsigned char smem[2 * BUF_BYTES];

    const int tid   = threadIdx.x;
    const int lane  = tid & 31;
    const int w     = tid >> 5;
    const int half  = lane >> 4;
    const int l16   = lane & 15;
    const int wr    = w & 3;
    const int wc    = w >> 2;
    const int tileN = blockIdx.x * 128;
    const int tileM = blockIdx.y * 128;

    const unsigned lds_base = (unsigned)(unsigned long long)(&smem[0]);

    const unsigned char* ga[4];
    const unsigned char* gb[4];
    unsigned la[4], lb[4];
#pragma unroll
    for (int j = 0; j < 4; ++j) {
        const int c   = j * 256 + tid;
        const int row = c >> 3;
        const int ch  = c & 7;
        ga[j] = qx8 + (size_t)(tileM + row) * CIN + ch * 16;
        gb[j] = qw8 + (size_t)(tileN + row) * CIN + ch * 16;
        la[j] = (unsigned)(row * LDS_ROW_STRIDE + ch * 16);
        lb[j] = (unsigned)(TILE_BYTES + row * LDS_ROW_STRIDE + ch * 16);
    }

    const unsigned char* As0 = smem + (size_t)(wr * 32 +  0 + l16) * LDS_ROW_STRIDE;
    const unsigned char* As1 = smem + (size_t)(wr * 32 + 16 + l16) * LDS_ROW_STRIDE;
    const unsigned char* Bs[4];
#pragma unroll
    for (int bi = 0; bi < 4; ++bi)
        Bs[bi] = smem + TILE_BYTES +
                 (size_t)(wc * 64 + bi * 16 + l16) * LDS_ROW_STRIDE;

    v8f c[2][4];
#pragma unroll
    for (int s = 0; s < 2; ++s)
#pragma unroll
        for (int bi = 0; bi < 4; ++bi) c[s][bi] = (v8f)(0.0f);

#pragma unroll
    for (int j = 0; j < 4; ++j) {
        async_copy_b128(lds_base + la[j], ga[j]);
        async_copy_b128(lds_base + lb[j], gb[j]);
    }
    wait_async0();
    __syncthreads();

    const int nsteps = CIN >> 7;
    for (int s = 0; s < nsteps; ++s) {
        const int      buf  = s & 1;
        const unsigned boff = (unsigned)(buf * BUF_BYTES);

        if (s + 1 < nsteps) {
            const int      k1  = (s + 1) << 7;
            const unsigned nbo = lds_base + (unsigned)((buf ^ 1) * BUF_BYTES);
#pragma unroll
            for (int j = 0; j < 4; ++j) {
                async_copy_b128(nbo + la[j], ga[j] + k1);
                async_copy_b128(nbo + lb[j], gb[j] + k1);
            }
        }

        v16i a0 = load_a_frag_lds(As0 + boff, half);
        v16i a1 = load_a_frag_lds(As1 + boff, half);
#pragma unroll
        for (int bi = 0; bi < 4; ++bi) {
            v16i b = load_a_frag_lds(Bs[bi] + boff, half);
            c[0][bi] = __builtin_amdgcn_wmma_f32_16x16x128_fp8_fp8(
                a0, b, (short)0, c[0][bi], false, false);
            c[1][bi] = __builtin_amdgcn_wmma_f32_16x16x128_fp8_fp8(
                a1, b, (short)0, c[1][bi], false, false);
        }

        wait_async0();
        __syncthreads();
    }

    const float scale = (accum[0] * inv_xn) * (accum[1] * inv_wn);

#pragma unroll
    for (int s = 0; s < 2; ++s) {
#pragma unroll
        for (int bi = 0; bi < 4; ++bi) {
            const int col = tileN + wc * 64 + bi * 16 + l16;
            const float bv = bias[col];
#pragma unroll
            for (int v = 0; v < 8; ++v) {
                const int row = tileM + wr * 32 + s * 16 + v + 8 * half;
                out[(size_t)row * COUT + col] = c[s][bi][v] * scale + bv;
            }
        }
    }
}

extern "C" void kernel_launch(void* const* d_in, const int* in_sizes, int n_in,
                              void* d_out, int out_size, void* d_ws, size_t ws_size,
                              hipStream_t stream) {
    const float* x    = (const float*)d_in[0];
    const float* wgt  = (const float*)d_in[1];
    const float* bias = (const float*)d_in[2];
    float*       out  = (float*)d_out;

    const int xsz  = in_sizes[0];
    const int wsz  = in_sizes[1];
    const int COUT = in_sizes[2];
    const int CIN  = wsz / COUT;
    const int N    = xsz / CIN;

    unsigned char* ws    = (unsigned char*)d_ws;
    float*         accum = (float*)ws;
    unsigned char* qx8   = ws + 256;
    unsigned char* qw8   = qx8 + (size_t)N * CIN;

    zero_accum_kernel<<<1, 64, 0, stream>>>(accum);

    const long long n4x = (long long)xsz / 4;
    const long long n4w = (long long)wsz / 4;
    quantize_abs_kernel<<<2048, 256, 0, stream>>>(x,   qx8, accum + 0, n4x);
    quantize_abs_kernel<<<2048, 256, 0, stream>>>(wgt, qw8, accum + 1, n4w);

    dim3 grid(COUT / 128, N / 128);
    xnor_gemm_fp8_wmma_kernel<<<grid, 256, 0, stream>>>(
        qx8, qw8, bias, out, accum, CIN, COUT,
        1.0f / (float)xsz, 1.0f / (float)wsz);
}
